// MoEClassical_38886633898787
// MI455X (gfx1250) — matrix operand order repaired and checked by reading, not compiled, not run on hardware
//
#include <hip/hip_runtime.h>
#include <hip/hip_bf16.h>

// ---------------------------------------------------------------------------
// MoE top-2 forward for MI455X (gfx1250, wave32, WMMA bf16 + TDM gather).
// Sizes fixed by the reference: C=1024, H=4096, E=8, N=B*T=8192, topk=2.
// ---------------------------------------------------------------------------

#define C_DIM   1024
#define H_DIM   4096
#define E_NUM   8
#define N_TOK   8192
#define HCHUNK  2048               // H processed in 2 chunks to bound LDS
#define XA_PAD  (C_DIM + 8)        // +8 bf16 = +16B: rotates LDS banks per row
#define HL_PAD  (HCHUNK + 8)

typedef __attribute__((ext_vector_type(16))) __bf16       v16bf;
typedef __attribute__((ext_vector_type(8)))  float        v8f;
typedef __attribute__((ext_vector_type(4)))  unsigned int v4u;
typedef __attribute__((ext_vector_type(8)))  int          v8i;
typedef __attribute__((ext_vector_type(4)))  int          v4i;

// ---- WMMA fragment helpers (contiguous-K layouts, k = 16*half + s, wave32) ----------

// A-matrix 16x32 bf16 frag.  lane&15 = row M; lanes 0-15 take K kb+0..15,
// lanes 16-31 take K kb+16..31.  Caller passes
// koff = kb + (laneHi ? 16 : 0).
__device__ __forceinline__ v16bf lds_a_frag(const __bf16* rowBase, int koff) {
  v16bf f;
#pragma unroll
  for (int i = 0; i < 16; ++i) {
    f[i]     = rowBase[koff + i];
  }
  return f;
}

// B-matrix 32x16 bf16 frag from an fp32 row in global memory.  lane&15 =
// column N; lanes 0-15 hold K kb+0..15, lanes 16-31 hold K kb+16..31.
// Caller passes koff = kb + (laneHi ? 16 : 0): 16 consecutive fp32 (64B).
__device__ __forceinline__ v16bf glb_b_frag(const float* __restrict__ rowBase,
                                            int koff) {
  const float4* p = reinterpret_cast<const float4*>(rowBase + koff);
  float4 q0 = p[0], q1 = p[1], q2 = p[2], q3 = p[3];
  v16bf f;
  f[0]=(__bf16)q0.x; f[1]=(__bf16)q0.y; f[2]=(__bf16)q0.z; f[3]=(__bf16)q0.w;
  f[4]=(__bf16)q1.x; f[5]=(__bf16)q1.y; f[6]=(__bf16)q1.z; f[7]=(__bf16)q1.w;
  f[8]=(__bf16)q2.x; f[9]=(__bf16)q2.y; f[10]=(__bf16)q2.z; f[11]=(__bf16)q2.w;
  f[12]=(__bf16)q3.x; f[13]=(__bf16)q3.y; f[14]=(__bf16)q3.z; f[15]=(__bf16)q3.w;
  return f;
}

// ---------------------------------------------------------------------------
// Kernel 1: router.  logits = x @ gate_w^T (raw logits to output tail),
// top-2 + renormalized softmax weights to workspace.
// ---------------------------------------------------------------------------
__global__ __launch_bounds__(256) void moe_router(
    const float* __restrict__ x, const float* __restrict__ gate_w,
    float* __restrict__ logits_out, int* __restrict__ tki,
    float* __restrict__ tkw) {
  const int n = blockIdx.x * blockDim.x + threadIdx.x;
  if (n >= N_TOK) return;
  const float4* xr = reinterpret_cast<const float4*>(x + (size_t)n * C_DIM);

  float lg[E_NUM];
#pragma unroll
  for (int e = 0; e < E_NUM; ++e) lg[e] = 0.0f;

  for (int c4 = 0; c4 < C_DIM / 4; ++c4) {
    float4 xv = xr[c4];
#pragma unroll
    for (int e = 0; e < E_NUM; ++e) {
      float4 gv = reinterpret_cast<const float4*>(gate_w + (size_t)e * C_DIM)[c4];
      lg[e] = fmaf(xv.x, gv.x, fmaf(xv.y, gv.y, fmaf(xv.z, gv.z, fmaf(xv.w, gv.w, lg[e]))));
    }
  }
#pragma unroll
  for (int e = 0; e < E_NUM; ++e) logits_out[(size_t)n * E_NUM + e] = lg[e];

  // top-2 (softmax is monotone, ties broken by lower index like lax.top_k)
  int i0 = 0;
#pragma unroll
  for (int e = 1; e < E_NUM; ++e) if (lg[e] > lg[i0]) i0 = e;
  int i1 = (i0 == 0) ? 1 : 0;
#pragma unroll
  for (int e = 0; e < E_NUM; ++e)
    if (e != i0 && lg[e] > lg[i1]) i1 = e;

  // normalized top-2 softmax weights == softmax over {lg[i0], lg[i1]}
  float e1 = __expf(lg[i1] - lg[i0]);
  float inv = 1.0f / (1.0f + e1);
  tki[2 * n]     = i0;  tkw[2 * n]     = inv;
  tki[2 * n + 1] = i1;  tkw[2 * n + 1] = e1 * inv;
}

// ---------------------------------------------------------------------------
// Kernel 2: deterministic per-expert compaction.  One block, 8 waves; wave e
// scans tokens in ascending order (-> monotonic lists, required by TDM gather
// OOB rules) and compacts via wave32 ballot prefix sums.
// ---------------------------------------------------------------------------
__global__ __launch_bounds__(256) void moe_compact(
    const int* __restrict__ tki, const float* __restrict__ tkw,
    int* __restrict__ counts, int* __restrict__ ltok,
    float* __restrict__ lw) {
  const int e    = threadIdx.x >> 5;   // wave id == expert id (wave32)
  const int lane = threadIdx.x & 31;
  int cnt = 0;
  for (int base = 0; base < N_TOK; base += 32) {
    const int n = base + lane;
    int slot = -1;
    if (tki[2 * n] == e)          slot = 0;
    else if (tki[2 * n + 1] == e) slot = 1;
    unsigned mask = (unsigned)__ballot(slot >= 0);
    if (slot >= 0) {
      int pos = cnt + __popc(mask & ((1u << lane) - 1u));
      ltok[e * N_TOK + pos] = n;
      lw  [e * N_TOK + pos] = tkw[2 * n + slot];
    }
    cnt += __popc(mask);
  }
  if (lane == 0) counts[e] = cnt;
}

// ---------------------------------------------------------------------------
// Kernel 3: zero the [N, C] output region (harness poisons d_out).
// ---------------------------------------------------------------------------
__global__ __launch_bounds__(256) void moe_zero(float4* __restrict__ out) {
  const size_t i = (size_t)blockIdx.x * blockDim.x + threadIdx.x;
  out[i] = make_float4(0.f, 0.f, 0.f, 0.f);
}

// ---------------------------------------------------------------------------
// Kernel 4: fused expert FFN.  One workgroup (8 wave32s) per 16-token tile of
// one expert's routed token list.  Grid is expert-major so the ~128 tiles of
// an expert reuse its 32 MB of fp32 weights out of the 192 MB L2; HBM reads
// the weight set ~once (the bandwidth floor, ~15us at 23.3 TB/s).
//   stage:  TDM gather-mode DMA pulls the 16 routed token rows (fp32) into
//           LDS (tensor_load_to_lds, TENSORcnt), then one LDS pass narrows
//           to a padded bf16 A-panel.
//   GEMM1:  h = relu(X Wfc^T)^2  (bf16 WMMA, f32 acc, h -> LDS bf16)
//   GEMM2:  Y = h Wproj^T        (bf16 WMMA, persistent f32 acc)
//   epilogue: out[tok] += w_tok * Y via fp32 atomicAdd (exactly 2
//   contributions per element -> commutative -> deterministic).
// ---------------------------------------------------------------------------
__global__ __launch_bounds__(256) void moe_expert_ffn(
    const float* __restrict__ x, const float* __restrict__ wfc,
    const float* __restrict__ wproj, const int* __restrict__ counts,
    const int* __restrict__ ltok, const float* __restrict__ lw,
    float* __restrict__ out) {
  // xf (fp32 TDM landing zone, 64KB) is dead before hl (66KB) is written:
  // overlap them to keep total LDS ~99KB.
  __shared__ union {
    float  xf[16][C_DIM];     // gathered token rows, fp32 (TDM target)
    __bf16 hl[16][HL_PAD];    // activation panel for current H chunk
  } u;
  __shared__ __bf16 xa[16][XA_PAD];   // bf16 A panel (~33 KB)
  __shared__ int    stok[16];
  __shared__ float  sw[16];

  const int e    = blockIdx.y;
  const int cnt  = counts[e];
  const int base = blockIdx.x * 16;
  if (base >= cnt) return;            // cnt > 0 past this point

  const int tid   = threadIdx.x;
  const int wave  = tid >> 5;
  const int lane  = tid & 31;
  const int laneN = lane & 15;
  const int hi    = lane >> 4;        // lane half selector

  if (tid < 16) {
    const int g = base + tid;
    if (g < cnt) { stok[tid] = ltok[e * N_TOK + g]; sw[tid] = lw[e * N_TOK + g]; }
    else         { stok[tid] = ltok[e * N_TOK + (cnt - 1)]; sw[tid] = 0.0f; }
  }

  // ---- TDM gather: 16 token rows of x -> u.xf (async, TENSORcnt) ----------
  if (wave == 0) {
    // Wave-uniform row indices (monotonic; tail repeats last valid index).
    int idx[16];
#pragma unroll
    for (int i = 0; i < 16; ++i) {
      const int g = base + i;
      idx[i] = ltok[e * N_TOK + (g < cnt ? g : cnt - 1)];
    }
    const unsigned lds_base =
        (unsigned)(size_t)(&u.xf[0][0]);              // LDS byte offset
    const unsigned long long ga = (unsigned long long)(size_t)x;

    // D# group 0: gather_mode=1, 16-bit indices, count=1, type=2.
    v4u g0 = { 0x80000001u, lds_base,
               (unsigned)(ga & 0xFFFFFFFFu),
               (unsigned)((ga >> 32) & 0x01FFFFFFu) | (2u << 30) };
    // D# group 1: data_size=4B; tensor = [N_TOK rows x C_DIM fp32],
    // tile = [16 rows x C_DIM]; row stride = C_DIM.
    v8i g1 = { (int)(2u << 16),                        // data_size=2 (4B)
               (int)((C_DIM & 0xFFFF) << 16),          // tensor_dim0 lo
               (int)((N_TOK & 0xFFFF) << 16),          // dim0 hi=0 | dim1 lo
               (int)((C_DIM & 0xFFFF) << 16),          // dim1 hi=0 | tile_dim0
               16,                                     // tile_dim1 = #indices
               C_DIM,                                  // tensor_dim0_stride lo
               0, 0 };
    // D# groups 2/3: sixteen 16-bit row indices.
    v4i g2 = { idx[0] | (idx[1] << 16), idx[2] | (idx[3] << 16),
               idx[4] | (idx[5] << 16), idx[6] | (idx[7] << 16) };
    v4i g3 = { idx[8]  | (idx[9]  << 16), idx[10] | (idx[11] << 16),
               idx[12] | (idx[13] << 16), idx[14] | (idx[15] << 16) };

#if __has_include(<hip/amd_detail/amd_gfx1250_TDM.h>)
    v8i zero8 = { 0, 0, 0, 0, 0, 0, 0, 0 };
    __builtin_amdgcn_tensor_load_to_lds(g0, g1, g2, g3, zero8, 0);
#else
    __builtin_amdgcn_tensor_load_to_lds(g0, g1, g2, g3, 0);
#endif
    __builtin_amdgcn_s_wait_tensorcnt(0);
  }
  __syncthreads();

  // Narrow gathered fp32 rows -> padded bf16 A panel (one LDS pass).
  for (int idx = tid; idx < 16 * (C_DIM / 4); idx += 256) {
    const int m  = idx / (C_DIM / 4);
    const int c  = (idx % (C_DIM / 4)) * 4;
    xa[m][c + 0] = (__bf16)u.xf[m][c + 0];
    xa[m][c + 1] = (__bf16)u.xf[m][c + 1];
    xa[m][c + 2] = (__bf16)u.xf[m][c + 2];
    xa[m][c + 3] = (__bf16)u.xf[m][c + 3];
  }
  __syncthreads();   // xf dead from here; u.hl may be written

  const float* wfc_e   = wfc   + (size_t)e * H_DIM * C_DIM;
  const float* wproj_e = wproj + (size_t)e * C_DIM * H_DIM;

  const v8f vzero = {0.f, 0.f, 0.f, 0.f, 0.f, 0.f, 0.f, 0.f};

  // Persistent GEMM2 accumulators: this wave owns output cols
  // [wave*128, wave*128+128) = 8 tiles of 16.
  v8f yacc[8];
#pragma unroll
  for (int t = 0; t < 8; ++t) yacc[t] = vzero;

  const int aoff1 = hi ? 16 : 0;   // A-frag K sub-offset
  const int boff1 = hi ? 16 : 0;   // B-frag K sub-offset

  for (int chunk = 0; chunk < H_DIM / HCHUNK; ++chunk) {
    const int hbase = chunk * HCHUNK;

    // ---- GEMM1: this wave produces h cols [hbase + wave*256, +256) --------
    for (int g = 0; g < 4; ++g) {
      v8f acc[4];
#pragma unroll
      for (int t = 0; t < 4; ++t) acc[t] = vzero;

      for (int k = 0; k < C_DIM / 32; ++k) {
        const int kb = k * 32;
        const v16bf a = lds_a_frag(&xa[laneN][0], kb + aoff1);
#pragma unroll
        for (int t = 0; t < 4; ++t) {
          const int h0 = hbase + wave * 256 + (g * 4 + t) * 16;
          const float* row = wfc_e + (size_t)(h0 + laneN) * C_DIM;
          const v16bf b = glb_b_frag(row, kb + boff1);
          acc[t] = __builtin_amdgcn_wmma_f32_16x16x32_bf16(
              false, a, false, b, (short)0, acc[t], false, false);
        }
      }
      // relu^2, narrow to bf16, store into h LDS (D layout: M=i+8*hi, N=lane&15)
#pragma unroll
      for (int t = 0; t < 4; ++t) {
        const int lcol = wave * 256 + (g * 4 + t) * 16 + laneN;
#pragma unroll
        for (int i = 0; i < 8; ++i) {
          const int m = i + (hi ? 8 : 0);
          float v = acc[t][i];
          v = (v > 0.0f) ? v * v : 0.0f;
          u.hl[m][lcol] = (__bf16)v;
        }
      }
    }
    __syncthreads();

    // ---- GEMM2 partial-K: accumulate over this chunk's 2048 h values ------
    for (int k = 0; k < HCHUNK / 32; ++k) {
      const int kb = k * 32;
      const v16bf a = lds_a_frag(&u.hl[laneN][0], kb + aoff1);
#pragma unroll
      for (int t = 0; t < 8; ++t) {
        const int c0 = wave * 128 + t * 16;
        const float* row = wproj_e + (size_t)(c0 + laneN) * H_DIM + hbase;
        const v16bf b = glb_b_frag(row, kb + boff1);
        yacc[t] = __builtin_amdgcn_wmma_f32_16x16x32_bf16(
            false, a, false, b, (short)0, yacc[t], false, false);
      }
    }
    __syncthreads();   // hl reused next chunk
  }

  // ---- epilogue: out[tok] += w_tok * Y ------------------------------------
#pragma unroll
  for (int t = 0; t < 8; ++t) {
    const int c = wave * 128 + t * 16 + laneN;
#pragma unroll
    for (int i = 0; i < 8; ++i) {
      const int m = i + (hi ? 8 : 0);
      const float v = yacc[t][i] * sw[m];
      atomicAdd(&out[(size_t)stok[m] * C_DIM + c], v);
    }
  }
}

// ---------------------------------------------------------------------------
// Host-side launch.  d_in: {x, gate_w, wfc, wproj, topk}.  d_out: out[N*C]
// then router_logits[N*E], fp32.
// ---------------------------------------------------------------------------
extern "C" void kernel_launch(void* const* d_in, const int* in_sizes, int n_in,
                              void* d_out, int out_size, void* d_ws,
                              size_t ws_size, hipStream_t stream) {
  const float* x      = (const float*)d_in[0];
  const float* gate_w = (const float*)d_in[1];
  const float* wfc    = (const float*)d_in[2];
  const float* wproj  = (const float*)d_in[3];

  float* out    = (float*)d_out;                          // [N, C]
  float* logits = (float*)d_out + (size_t)N_TOK * C_DIM;  // [N, E]

  // Workspace layout (~640 KB total)
  char*  w      = (char*)d_ws;
  int*   counts = (int*)w;                                   // 8 ints
  int*   tki    = (int*)(w + 256);                           // 2N ints
  float* tkw    = (float*)(w + 256 + 2 * N_TOK * 4);         // 2N floats
  int*   ltok   = (int*)(w + 256 + 4 * N_TOK * 4);           // E*N ints
  float* lw     = (float*)(w + 256 + 4 * N_TOK * 4 + E_NUM * N_TOK * 4);

  moe_router<<<N_TOK / 256, 256, 0, stream>>>(x, gate_w, logits, tki, tkw);
  moe_compact<<<1, 256, 0, stream>>>(tki, tkw, counts, ltok, lw);
  moe_zero<<<(N_TOK * C_DIM / 4) / 256, 256, 0, stream>>>((float4*)out);

  dim3 grid(N_TOK / 16, E_NUM);   // expert-major tiles; early-exit past count
  moe_expert_ffn<<<grid, 256, 0, stream>>>(x, wfc, wproj, counts, ltok, lw, out);
}
